// Net_57423712747598
// MI455X (gfx1250) — matrix operand order repaired and checked by reading, not compiled, not run on hardware
//
#include <hip/hip_runtime.h>
#include <hip/hip_bf16.h>
#include <math.h>
#include <stdint.h>

typedef __attribute__((ext_vector_type(16))) _Float16 v16h;
typedef __attribute__((ext_vector_type(8)))  _Float16 h8;
typedef __attribute__((ext_vector_type(8)))  float    v8f;

static __device__ __forceinline__ float siluf(float x){ return x / (1.f + __expf(-x)); }
static __device__ __forceinline__ float sigmf(float x){ return 1.f / (1.f + __expf(-x)); }
static __device__ __forceinline__ float lreluf(float x){ return fmaxf(x, 0.2f*x); }
static __device__ __forceinline__ float softplusf(float x){ return (x > 20.f) ? x : log1pf(__expf(x)); }

static __device__ __forceinline__ v8f wmma16(v16h a, v16h b, v8f c){
  return __builtin_amdgcn_wmma_f32_16x16x32_f16(false, a, false, b, (short)0, c, false, false);
}
static __device__ __forceinline__ v16h cat8(h8 lo, h8 hi){
  return __builtin_shufflevector(lo, hi, 0,1,2,3,4,5,6,7,8,9,10,11,12,13,14,15);
}
static __device__ __forceinline__ h8 cvt8(float4 a, float4 b){
  h8 r;
  r[0]=(_Float16)a.x; r[1]=(_Float16)a.y; r[2]=(_Float16)a.z; r[3]=(_Float16)a.w;
  r[4]=(_Float16)b.x; r[5]=(_Float16)b.y; r[6]=(_Float16)b.z; r[7]=(_Float16)b.w;
  return r;
}

// ---------------------------------------------------------------------------
// Weight pack kernels: f32 -> f16, fragment-linear, zero padded.
// GEMM: Wp[(ch*Npad + n)*32 + kk], kk = K-index within chunk.
//   wt==0: W is [K][N] row-major; wt==1: W is [N][K].
// ---------------------------------------------------------------------------
__global__ void packw_kernel(const float* __restrict__ W, _Float16* __restrict__ Wp,
                             int K, int N, int Npad, int nch, int wt){
  int tot = nch*Npad*32;
  int i = blockIdx.x*blockDim.x + threadIdx.x; if (i >= tot) return;
  int kk = i & 31; int n = (i >> 5) % Npad; int ch = i / (32*Npad);
  int k = ch*32 + kk;
  float v = 0.f;
  if (k < K && n < N) v = wt ? W[(size_t)n*K + k] : W[(size_t)k*N + n];
  Wp[i] = (_Float16)v;
}
// Conv: Wp[((tap*nc + ch)*32 + n)*32 + kk]; w is [Cout][Cin][3][3]
__global__ void packconv_kernel(const float* __restrict__ w, _Float16* __restrict__ Wp,
                                int Cin, int Cout, int nc){
  int tot = 9*nc*32*32;
  int i = blockIdx.x*blockDim.x + threadIdx.x; if (i >= tot) return;
  int kk = i & 31; int n = (i >> 5) & 31; int ch = (i >> 10) % nc; int tap = i / (1024*nc);
  int c = ch*32 + kk;
  float v = 0.f;
  if (c < Cin && n < Cout) v = w[((size_t)n*Cin + c)*9 + tap];
  Wp[i] = (_Float16)v;
}

// ---------------------------------------------------------------------------
// Token GEMM: C[M,N] = act(A[M,K] @ W + bias), packed f16 weights, branch-free
// inner loop. Kpad multiple of 32 (A rows must have >= Kpad readable floats).
// act: 0 none, 1 silu, 2 softplus. One wave per 16x16 tile.
// ---------------------------------------------------------------------------
__global__ void gemm_wmma_kernel(const float* __restrict__ A, const _Float16* __restrict__ Wp,
                                 const float* __restrict__ bias, float* __restrict__ C,
                                 int M, int N, int Kpad, int lda, int ldc, int act, int Npad)
{
  int lane = threadIdx.x & 31;
  int m0 = blockIdx.x * 16, n0 = blockIdx.y * 16;
  int half = lane >> 4, mr = lane & 15;
  int n = n0 + mr;
  v8f acc = {};
  const float* Arow = A + (size_t)(m0 + mr) * lda;
  int nch = Kpad >> 5;
  for (int ch = 0; ch < nch; ++ch) {
    int k0 = ch * 32;
    const float* pa = Arow + k0 + 8*half;
    if (ch + 1 < nch) __builtin_prefetch(Arow + k0 + 32, 0, 1);
    float4 a0 = *(const float4*)(pa);
    float4 a1 = *(const float4*)(pa + 4);
    float4 a2 = *(const float4*)(pa + 16);
    float4 a3 = *(const float4*)(pa + 20);
    v16h af = cat8(cvt8(a0, a1), cvt8(a2, a3));
    const _Float16* pw = Wp + ((size_t)ch*Npad + n)*32 + 8*half;
    v16h bf = cat8(*(const h8*)pw, *(const h8*)(pw + 16));
    acc = wmma16(af, bf, acc);
  }
  if (n < N) {
    float bv = bias ? bias[n] : 0.f;
#pragma unroll
    for (int v = 0; v < 8; ++v) {
      int m = m0 + v + 8*half;
      float x = acc[v] + bv;
      if (act == 1) x = siluf(x);
      else if (act == 2) x = softplusf(x);
      C[(size_t)m*ldc + n] = x;
    }
  }
}

// ---------------------------------------------------------------------------
// 3x3 SAME conv, NCHW f32, implicit GEMM via WMMA, packed f16 weights.
// Block = 64 threads (2 waves); wave w -> out channels [16w, 16w+16).
// Tile = 16 pixels of one row. act: 0 none, 3 lrelu, 4 relu.
// out = act(conv + bias) + (res ? res : 0)
// ---------------------------------------------------------------------------
__global__ void conv3x3_wmma_kernel(const float* __restrict__ in, const _Float16* __restrict__ wp,
                                    const float* __restrict__ bias, const float* __restrict__ res,
                                    float* __restrict__ out,
                                    int Cin, int Cout, int H, int W, int act)
{
  __shared__ __align__(16) _Float16 lds[3 * 18 * 64];
  int x0 = blockIdx.x * 16, y = blockIdx.y, bz = blockIdx.z;
  int tid = threadIdx.x;
  int cplog2 = (Cin > 32) ? 6 : 5;
  int Cpad = 1 << cplog2;
  int nc = Cpad >> 5;
  // stage zero-padded input patch: rows y-1..y+1, cols x0-1..x0+16, ch 0..Cpad-1
  int tot = 3 * 18 * Cpad;
  for (int e = tid; e < tot; e += 64) {
    int cc = e % 18; int t = e / 18;
    int c = t & (Cpad - 1); int r = t >> cplog2;
    int yy = y + r - 1, xx = x0 + cc - 1;
    float v = 0.f;
    if (c < Cin && yy >= 0 && yy < H && xx >= 0 && xx < W)
      v = in[((size_t)(bz*Cin + c)*H + yy)*W + xx];
    lds[(r*18 + cc)*64 + c] = (_Float16)v;
  }
  __syncthreads();

  int wave = tid >> 5, lane = tid & 31;
  int half = lane >> 4, mr = lane & 15;
  int n = wave*16 + mr;            // out channel
  v8f acc = {};
  for (int tap = 0; tap < 9; ++tap) {
    int dy = tap / 3, dx = tap % 3;
    const _Float16* lrow = &lds[(dy*18 + (mr + dx))*64];
    for (int ch = 0; ch < nc; ++ch) {
      const _Float16* pa = lrow + ch*32 + 8*half;
      v16h af = cat8(*(const h8*)pa, *(const h8*)(pa + 16));
      const _Float16* pb = wp + (((size_t)(tap*nc + ch)*32 + n)*32) + 8*half;
      v16h bf = cat8(*(const h8*)pb, *(const h8*)(pb + 16));
      acc = wmma16(af, bf, acc);
    }
  }
  if (n < Cout) {
    float bv = bias ? bias[n] : 0.f;
#pragma unroll
    for (int v = 0; v < 8; ++v) {
      int px = x0 + v + 8*half;
      float xv = acc[v] + bv;
      if (act == 3) xv = lreluf(xv);
      else if (act == 4) xv = fmaxf(xv, 0.f);
      size_t o = ((size_t)(bz*Cout + n)*H + y)*W + px;
      if (res) xv += res[o];
      out[o] = xv;
    }
  }
}

// ---------------------------------------------------------------------------
// Elementwise / small kernels
// ---------------------------------------------------------------------------
__global__ void fill0_kernel(float* p, int n){
  int i = blockIdx.x*blockDim.x + threadIdx.x; if (i < n) p[i] = 0.f;
}
__global__ void addto_kernel(float* dst, const float* src, int n){
  int i = blockIdx.x*blockDim.x + threadIdx.x; if (i < n) dst[i] += src[i];
}
__global__ void add3_kernel(const float* a, const float* b, const float* c, float* o, int n){
  int i = blockIdx.x*blockDim.x + threadIdx.x; if (i < n) o[i] = a[i] + b[i] + c[i];
}
__global__ void up_kernel(const float* src, float* dst, int Bn, int C, int Hs, int Ws, int r, int add){
  int Hd = Hs*r, Wd = Ws*r;
  int tot = Bn*C*Hd*Wd;
  int i = blockIdx.x*blockDim.x + threadIdx.x; if (i >= tot) return;
  int x = i % Wd; int y = (i/Wd) % Hd; int c = (i/(Wd*Hd)) % C; int b = i/(Wd*Hd*C);
  float v = src[((size_t)(b*C + c)*Hs + y/r)*Ws + x/r];
  if (add) dst[i] += v; else dst[i] = v;
}
__global__ void dwt_kernel(const float* x, float* LL, float* HH, float* HL, float* LH,
                           int Bn, int C, int H, int W){
  int Ho = H/2, Wo = W/2;
  int tot = Bn*C*Ho*Wo;
  int i = blockIdx.x*blockDim.x + threadIdx.x; if (i >= tot) return;
  int xo = i % Wo; int yo = (i/Wo) % Ho; int c = (i/(Wo*Ho)) % C; int b = i/(Wo*Ho*C);
  const float* p = x + (size_t)(b*C + c)*H*W;
  float x1 = p[(2*yo)*W   + 2*xo  ] * 0.5f;
  float x2 = p[(2*yo+1)*W + 2*xo  ] * 0.5f;
  float x3 = p[(2*yo)*W   + 2*xo+1] * 0.5f;
  float x4 = p[(2*yo+1)*W + 2*xo+1] * 0.5f;
  LL[i] =  x1 + x2 + x3 + x4;
  HH[i] =  x1 - x2 - x3 + x4;
  HL[i] = -x1 - x2 + x3 + x4;
  LH[i] = -x1 + x2 - x3 + x4;
}
__global__ void copych_kernel(const float* src, float* dst, int Bn, int Csrc, int Cdst, int coff, int HW){
  int tot = Bn*Csrc*HW;
  int i = blockIdx.x*blockDim.x + threadIdx.x; if (i >= tot) return;
  int p = i % HW; int c = (i/HW) % Csrc; int b = i/(HW*Csrc);
  dst[((size_t)(b*Cdst + coff + c))*HW + p] = src[i];
}
__global__ void nchw2tok_kernel(const float* x, float* t, int Bn, int C, int HW){
  int tot = Bn*C*HW;
  int i = blockIdx.x*blockDim.x + threadIdx.x; if (i >= tot) return;
  int p = i % HW; int c = (i/HW) % C; int b = i/(HW*C);
  t[((size_t)b*HW + p)*C + c] = x[i];
}
__global__ void tok2nchw_kernel(const float* t, float* x, int Bn, int C, int HW){
  int tot = Bn*C*HW;
  int i = blockIdx.x*blockDim.x + threadIdx.x; if (i >= tot) return;
  int p = i % HW; int c = (i/HW) % C; int b = i/(HW*C);
  x[i] = t[((size_t)b*HW + p)*C + c];
}
__global__ void instnorm_kernel(float* x, const float* w, const float* b, int C, int HW){
  int c = blockIdx.x, bb = blockIdx.y;
  float* p = x + ((size_t)bb*C + c)*HW;
  __shared__ float s1[256], s2[256];
  int tid = threadIdx.x;
  float a = 0.f, q = 0.f;
  for (int i = tid; i < HW; i += 256){ float v = p[i]; a += v; q += v*v; }
  s1[tid] = a; s2[tid] = q; __syncthreads();
  for (int s = 128; s > 0; s >>= 1){
    if (tid < s){ s1[tid] += s1[tid+s]; s2[tid] += s2[tid+s]; }
    __syncthreads();
  }
  float mu = s1[0]/HW; float var = s2[0]/HW - mu*mu;
  float inv = rsqrtf(var + 1e-5f);
  float ww = w[c], bv = b[c];
  for (int i = tid; i < HW; i += 256) p[i] = (p[i]-mu)*inv*ww + bv;
}
__global__ void ln_kernel(const float* x, const float* w, const float* b, float* y, int T){
  int t = blockIdx.x*blockDim.x + threadIdx.x; if (t >= T) return;
  const float* p = x + (size_t)t*32; float* q = y + (size_t)t*32;
  float mu = 0.f;
  for (int c = 0; c < 32; ++c) mu += p[c];
  mu *= (1.f/32.f);
  float var = 0.f;
  for (int c = 0; c < 32; ++c){ float d = p[c]-mu; var += d*d; }
  var *= (1.f/32.f);
  float inv = rsqrtf(var + 1e-5f);
  for (int c = 0; c < 32; ++c) q[c] = (p[c]-mu)*inv*w[c] + b[c];
}
__global__ void dwconv1d_silu_kernel(const float* x, int ldx, const float* w, const float* bias,
                                     float* y, int Bn, int L, int D){
  int tot = Bn*L*D;
  int i = blockIdx.x*blockDim.x + threadIdx.x; if (i >= tot) return;
  int d = i % D; int l = (i/D) % L; int b = i/(D*L);
  float s = bias[d];
#pragma unroll
  for (int j = 0; j < 4; ++j){
    int l2 = l - 3 + j;
    if (l2 >= 0) s += w[d*4 + j] * x[((size_t)b*L + l2)*ldx + d];
  }
  y[((size_t)b*L + l)*D + d] = siluf(s);
}
__global__ void gate_mul_kernel(const float* y, const float* xz, float* g, int Bn, int L){
  int tot = Bn*L*64;
  int i = blockIdx.x*blockDim.x + threadIdx.x; if (i >= tot) return;
  int d = i % 64; size_t bl = (size_t)i / 64;
  float z = xz[bl*128 + 64 + d];
  g[i] = y[i] * siluf(z);
}
__global__ void dw3x3_kernel(const float* x, const float* w, const float* bias, float* out,
                             int Bn, int C, int H, int W){
  int tot = Bn*C*H*W;
  int i = blockIdx.x*blockDim.x + threadIdx.x; if (i >= tot) return;
  int xx = i % W; int yy = (i/W) % H; int c = (i/(W*H)) % C; int b = i/(W*H*C);
  const float* p = x + (size_t)(b*C + c)*H*W;
  float s = bias[c];
  for (int dy = 0; dy < 3; ++dy)
    for (int dx = 0; dx < 3; ++dx){
      int y2 = yy + dy - 1, x2 = xx + dx - 1;
      if (y2 >= 0 && y2 < H && x2 >= 0 && x2 < W)
        s += w[c*9 + dy*3 + dx] * p[y2*W + x2];
    }
  out[i] = s + x[i];
}
__global__ void meanhw_kernel(const float* x, float* y, int C, int HW){
  int c = blockIdx.x, b = blockIdx.y;
  const float* p = x + ((size_t)b*C + c)*HW;
  __shared__ float s[256];
  int tid = threadIdx.x;
  float a = 0.f;
  for (int i = tid; i < HW; i += 256) a += p[i];
  s[tid] = a; __syncthreads();
  for (int r = 128; r > 0; r >>= 1){ if (tid < r) s[tid] += s[tid+r]; __syncthreads(); }
  if (tid == 0) y[b*C + c] = s[0] / HW;
}
__global__ void fc_kernel(const float* x, const float* w, const float* b, float* y,
                          int Bn, int Cin, int Cout, int act){
  int i = blockIdx.x*blockDim.x + threadIdx.x; if (i >= Bn*Cout) return;
  int o = i % Cout; int bb = i / Cout;
  float s = b[o];
  for (int ci = 0; ci < Cin; ++ci) s += w[o*Cin + ci] * x[bb*Cin + ci];
  if (act == 4) s = fmaxf(s, 0.f);
  else if (act == 5) s = sigmf(s);
  y[i] = s;
}
__global__ void scale_add_kernel(float* dst, const float* res, const float* att, int Bn, int C, int HW){
  int tot = Bn*C*HW;
  int i = blockIdx.x*blockDim.x + threadIdx.x; if (i >= tot) return;
  int c = (i/HW) % C; int b = i/(HW*C);
  dst[i] += att[b*C + c] * res[i];
}

// ---------------------------------------------------------------------------
// Chunked selective scan: L=4096 -> 32 chunks of 128.
// dbl layout: (b,l,48): cols 2..17 = B, 18..33 = C.
// ---------------------------------------------------------------------------
#define NCHUNK 32
#define CLEN 128
__global__ void scanA_kernel(const float* u, const float* dt, const float* dbl, const float* Alog,
                             float* Pm, float* Hl, int Bn, int L){
  int t = blockIdx.x*blockDim.x + threadIdx.x;
  int tot = Bn*64*NCHUNK; if (t >= tot) return;
  int ch = t % NCHUNK; int d = (t/NCHUNK) % 64; int b = t/(NCHUNK*64);
  float Av[16], h[16], p[16];
#pragma unroll
  for (int n = 0; n < 16; ++n){ Av[n] = -__expf(Alog[d*16 + n]); h[n] = 0.f; p[n] = 1.f; }
  for (int i = 0; i < CLEN; ++i){
    size_t bl = (size_t)b*L + ch*CLEN + i;
    float dtv = dt[bl*64 + d];
    float uv  = u[bl*64 + d];
    const float* db = dbl + bl*48;
#pragma unroll
    for (int n = 0; n < 16; ++n){
      float e = __expf(dtv*Av[n]);
      h[n] = e*h[n] + dtv*db[2 + n]*uv;
      p[n] *= e;
    }
  }
  size_t o = ((size_t)(b*64 + d)*NCHUNK + ch)*16;
#pragma unroll
  for (int n = 0; n < 16; ++n){ Pm[o+n] = p[n]; Hl[o+n] = h[n]; }
}
__global__ void scanB_kernel(const float* Pm, const float* Hl, float* Hi, int Bn){
  int t = blockIdx.x*blockDim.x + threadIdx.x;
  if (t >= Bn*64) return;
  float h[16];
#pragma unroll
  for (int n = 0; n < 16; ++n) h[n] = 0.f;
  for (int ch = 0; ch < NCHUNK; ++ch){
    size_t o = ((size_t)t*NCHUNK + ch)*16;
#pragma unroll
    for (int n = 0; n < 16; ++n){
      Hi[o+n] = h[n];
      h[n] = Pm[o+n]*h[n] + Hl[o+n];
    }
  }
}
__global__ void scanC_kernel(const float* u, const float* dt, const float* dbl, const float* Alog,
                             const float* Dp, const float* Hi, float* y, int Bn, int L){
  int t = blockIdx.x*blockDim.x + threadIdx.x;
  int tot = Bn*64*NCHUNK; if (t >= tot) return;
  int ch = t % NCHUNK; int d = (t/NCHUNK) % 64; int b = t/(NCHUNK*64);
  float Av[16], h[16];
  size_t o = ((size_t)(b*64 + d)*NCHUNK + ch)*16;
#pragma unroll
  for (int n = 0; n < 16; ++n){ Av[n] = -__expf(Alog[d*16 + n]); h[n] = Hi[o+n]; }
  float Dv = Dp[d];
  for (int i = 0; i < CLEN; ++i){
    size_t bl = (size_t)b*L + ch*CLEN + i;
    float dtv = dt[bl*64 + d];
    float uv  = u[bl*64 + d];
    const float* db = dbl + bl*48;
    float yv = 0.f;
#pragma unroll
    for (int n = 0; n < 16; ++n){
      float e = __expf(dtv*Av[n]);
      h[n] = e*h[n] + dtv*db[2 + n]*uv;
      yv += h[n]*db[18 + n];
    }
    y[bl*64 + d] = yv + uv*Dv;
  }
}

// ---------------------------------------------------------------------------
// Host orchestration
// ---------------------------------------------------------------------------
struct MambaP { const float *Alog,*Dp,*Wdt,*Win,*Win2,*Wout,*Wx,*bdt,*conv2b,*conv2w,*convb,*convw; };
struct SingleP { MambaP m; const float *nb,*nw; };
struct CrossP { const float *dwb,*dww; MambaP m; const float *n1b,*n1w,*n2b,*n2w; };
struct HinP { const float *c1b,*c1w,*c2b,*c2w,*inb,*inw; };
struct EncP { const float *b; HinP blk[3]; const float *w; };

extern "C" void kernel_launch(void* const* d_in, const int* in_sizes, int n_in,
                              void* d_out, int out_size, void* d_ws, size_t ws_size,
                              hipStream_t stream)
{
  (void)in_sizes; (void)n_in; (void)out_size; (void)ws_size;
  const float* hs  = (const float*)d_in[0];
  const float* pan = (const float*)d_in[1];
  int cur = 2;
  auto nxt = [&]() -> const float* { return (const float*)d_in[cur++]; };

  // ---- unpack params in jax pytree order (alphabetical dict keys) ----
  auto rdMamba = [&](bool cross){
    MambaP m{};
    m.Alog = nxt(); m.Dp = nxt(); m.Wdt = nxt(); m.Win = nxt();
    if (cross) m.Win2 = nxt();
    m.Wout = nxt(); m.Wx = nxt(); m.bdt = nxt();
    if (cross){ m.conv2b = nxt(); m.conv2w = nxt(); }
    m.convb = nxt(); m.convw = nxt();
    return m;
  };
  auto rdEnc = [&](){
    EncP e{};
    e.b = nxt();
    for (int i = 0; i < 3; ++i){
      e.blk[i].c1b = nxt(); e.blk[i].c1w = nxt();
      e.blk[i].c2b = nxt(); e.blk[i].c2w = nxt();
      e.blk[i].inb = nxt(); e.blk[i].inw = nxt();
    }
    e.w = nxt();
    return e;
  };
  CrossP df[5];
  for (int i = 0; i < 5; ++i){
    df[i].dwb = nxt(); df[i].dww = nxt();
    df[i].m = rdMamba(true);
    df[i].n1b = nxt(); df[i].n1w = nxt(); df[i].n2b = nxt(); df[i].n2w = nxt();
  }
  EncP hs_enc  = rdEnc();
  EncP hs_enc1 = rdEnc();
  const float* hs_tokb = nxt(); const float* hs_tokw = nxt();
  SingleP ms_fe[8];
  for (int i = 0; i < 8; ++i){ ms_fe[i].m = rdMamba(false); ms_fe[i].nb = nxt(); ms_fe[i].nw = nxt(); }
  const float *cib = nxt(), *ciw = nxt(), *clb = nxt(), *clw = nxt();
  const float *d1b = nxt(), *d1w = nxt(), *d2b = nxt(), *d2w = nxt();
  const float *p1b = nxt(), *p1w = nxt(), *p2b = nxt(), *p2w = nxt();
  EncP pan_enc  = rdEnc();
  EncP pan_enc1 = rdEnc();
  SingleP pan_fe[8];
  for (int i = 0; i < 8; ++i){ pan_fe[i].m = rdMamba(false); pan_fe[i].nb = nxt(); pan_fe[i].nw = nxt(); }
  const float* pan_tokb = nxt(); const float* pan_tokw = nxt();
  const float *sf1b = nxt(), *sf1w = nxt(), *sf2b = nxt(), *sf2w = nxt();

  // ---- workspace carve ----
  float* wp = (float*)d_ws;
  auto alloc = [&](size_t n){ float* r = wp; wp += n; return r; };
  const int Bn = 2, HW = 4096, L = 4096, MT = 8192;  // MT = Bn*L tokens
  float* hs_bic = alloc(2*31*64*64);
  float* hsLL = alloc(63488); float* hsHH = alloc(63488); float* hsHL = alloc(63488); float* hsLH = alloc(63488);
  float* panLL = alloc(2048); float* panHH = alloc(2048); float* panHL = alloc(2048); float* panLH = alloc(2048);
  float* hs_f = alloc(262144); float* pan_f = alloc(262144);
  float* ll_f = alloc(65536);  float* hf_f = alloc(65536);
  float* t0 = alloc(262144);   float* t1 = alloc(262144);
  float* cat = alloc(524288);
  float* hs_t = alloc(262144); float* pan_t = alloc(262144);
  float* resi = alloc(262144);
  float* tka = alloc(262144);  float* tkb = alloc(262144); float* tkc = alloc(262144);
  float* xz  = alloc(1048576);
  float* xcc = alloc(524288);
  float* x2l = alloc(524288);  float* x2c = alloc(524288);
  float* dbl = alloc(393216);  // (b,l,48)
  float* dtb = alloc(524288);
  float* ysc = alloc(524288);
  float* gbuf= alloc(524288);
  float* scP = alloc(65536);   float* scH = alloc(65536); float* scI = alloc(65536);
  float* ym  = alloc(64);      float* z1b = alloc(16);    float* zatt = alloc(64);
  // packed weight scratch (reused serially; stream ordering guarantees safety)
  _Float16* hp = (_Float16*)(void*)((((uintptr_t)wp) + 15) & ~(uintptr_t)15);
  _Float16* gwp = hp; hp += 8192;    // gemm pack scratch (max 2*48*32=3072, Win 1*128*32=4096)
  _Float16* cwp = hp;                // conv pack scratch (max 9*2*32*32=18432)

  // ---- launch helpers ----
  auto gl = [](int n){ return dim3((unsigned)((n + 255) / 256)); };
  auto conv3 = [&](const float* in, const float* w, const float* b, const float* res,
                   float* out, int Cin, int Cout, int H, int W, int act){
    int nc = (Cin > 32) ? 2 : 1;
    packconv_kernel<<<gl(9*nc*32*32), 256, 0, stream>>>(w, cwp, Cin, Cout, nc);
    conv3x3_wmma_kernel<<<dim3(W/16, H, Bn), 64, 0, stream>>>(in, cwp, b, res, out, Cin, Cout, H, W, act);
  };
  auto gemm = [&](const float* A, const float* Wm, const float* bias, float* C,
                  int M, int N, int K, int lda, int ldc, int act, int wt){
    int nch = (K + 31) / 32, Npad = (N + 15) & ~15;
    packw_kernel<<<gl(nch*Npad*32), 256, 0, stream>>>(Wm, gwp, K, N, Npad, nch, wt);
    gemm_wmma_kernel<<<dim3(M/16, Npad/16), 32, 0, stream>>>(A, gwp, bias, C, M, N, nch*32, lda, ldc, act, Npad);
  };
  auto encoder = [&](const float* in, int Cin, const EncP& e, float* out, float* tmp, int H, int W){
    conv3(in, e.w, e.b, nullptr, out, Cin, 32, H, W, 0);
    for (int i = 0; i < 3; ++i){
      const HinP& hb = e.blk[i];
      conv3(out, hb.c1w, hb.c1b, nullptr, tmp, 32, 32, H, W, 3);
      instnorm_kernel<<<dim3(16, Bn), 256, 0, stream>>>(tmp, hb.inw, hb.inb, 32, H*W);
      conv3(tmp, hb.c2w, hb.c2b, out, out, 32, 32, H, W, 3);
    }
  };
  auto run_scan = [&](const float* u, const MambaP& m){
    scanA_kernel<<<gl(Bn*64*NCHUNK), 256, 0, stream>>>(u, dtb, dbl, m.Alog, scP, scH, Bn, L);
    scanB_kernel<<<1, 128, 0, stream>>>(scP, scH, scI, Bn);
    scanC_kernel<<<gl(Bn*64*NCHUNK), 256, 0, stream>>>(u, dtb, dbl, m.Alog, m.Dp, scI, ysc, Bn, L);
  };
  auto mamba_core = [&](const MambaP& m, const float* xin, const float* u_ovr, float* xout){
    gemm(xin, m.Win, nullptr, xz, MT, 128, 32, 32, 128, 0, 0);
    dwconv1d_silu_kernel<<<gl(Bn*L*64), 256, 0, stream>>>(xz, 128, m.convw, m.convb, xcc, Bn, L, 64);
    gemm(xcc, m.Wx, nullptr, dbl, MT, 34, 64, 64, 48, 0, 0);
    gemm(dbl, m.Wdt, m.bdt, dtb, MT, 64, 2, 48, 64, 2, 0);
    run_scan(u_ovr ? u_ovr : xcc, m);
    gate_mul_kernel<<<gl(Bn*L*64), 256, 0, stream>>>(ysc, xz, gbuf, Bn, L);
    gemm(gbuf, m.Wout, nullptr, xout, MT, 32, 64, 64, 32, 0, 0);
  };

  // ---- forward ----
  up_kernel<<<gl(2*31*64*64), 256, 0, stream>>>(hs, hs_bic, Bn, 31, 16, 16, 4, 0);
  dwt_kernel<<<gl(2*31*1024), 256, 0, stream>>>(hs_bic, hsLL, hsHH, hsHL, hsLH, Bn, 31, 64, 64);
  dwt_kernel<<<gl(2*1*1024), 256, 0, stream>>>(pan, panLL, panHH, panHL, panLH, Bn, 1, 64, 64);

  encoder(hs_bic, 31, hs_enc1, hs_f, t0, 64, 64);
  encoder(pan, 1, pan_enc1, pan_f, t0, 64, 64);

  copych_kernel<<<gl(2*31*1024), 256, 0, stream>>>(hsLL, cat, Bn, 31, 32, 0, 1024);
  copych_kernel<<<gl(2*1*1024), 256, 0, stream>>>(panLL, cat, Bn, 1, 32, 31, 1024);
  encoder(cat, 32, hs_enc, ll_f, t0, 32, 32);

  add3_kernel<<<gl(2*31*1024), 256, 0, stream>>>(hsHL, hsLH, hsHH, t1, 2*31*1024);
  encoder(t1, 31, pan_enc, hf_f, t0, 32, 32);

  up_kernel<<<gl(2*32*64*64), 256, 0, stream>>>(ll_f, hs_f, Bn, 32, 32, 32, 2, 1);
  up_kernel<<<gl(2*32*64*64), 256, 0, stream>>>(hf_f, pan_f, Bn, 32, 32, 32, 2, 1);

  copych_kernel<<<gl(2*32*HW), 256, 0, stream>>>(hs_f, cat, Bn, 32, 64, 0, HW);
  copych_kernel<<<gl(2*32*HW), 256, 0, stream>>>(pan_f, cat, Bn, 32, 64, 32, HW);
  conv3(cat, sf1w, sf1b, hs_f, hs_f, 64, 32, 64, 64, 0);
  copych_kernel<<<gl(2*32*HW), 256, 0, stream>>>(pan_f, cat, Bn, 32, 64, 0, HW);
  copych_kernel<<<gl(2*32*HW), 256, 0, stream>>>(hs_f, cat, Bn, 32, 64, 32, HW);
  conv3(cat, sf2w, sf2b, pan_f, pan_f, 64, 32, 64, 64, 0);

  nchw2tok_kernel<<<gl(2*32*HW), 256, 0, stream>>>(hs_f, tka, Bn, 32, HW);
  gemm(tka, hs_tokw, hs_tokb, hs_t, MT, 32, 32, 32, 32, 0, 1);
  nchw2tok_kernel<<<gl(2*32*HW), 256, 0, stream>>>(pan_f, tka, Bn, 32, HW);
  gemm(tka, pan_tokw, pan_tokb, pan_t, MT, 32, 32, 32, 32, 0, 1);

  const int NT = 262144;
  fill0_kernel<<<gl(NT), 256, 0, stream>>>(resi, NT);
  for (int i = 0; i < 8; ++i){
    addto_kernel<<<gl(NT), 256, 0, stream>>>(resi, hs_t, NT);
    ln_kernel<<<gl(MT), 256, 0, stream>>>(resi, ms_fe[i].nw, ms_fe[i].nb, tka, MT);
    mamba_core(ms_fe[i].m, tka, nullptr, hs_t);
  }
  fill0_kernel<<<gl(NT), 256, 0, stream>>>(resi, NT);
  for (int i = 0; i < 8; ++i){
    addto_kernel<<<gl(NT), 256, 0, stream>>>(resi, pan_t, NT);
    ln_kernel<<<gl(MT), 256, 0, stream>>>(resi, pan_fe[i].nw, pan_fe[i].nb, tka, MT);
    mamba_core(pan_fe[i].m, tka, nullptr, pan_t);
  }
  fill0_kernel<<<gl(NT), 256, 0, stream>>>(resi, NT);
  for (int i = 0; i < 5; ++i){
    addto_kernel<<<gl(NT), 256, 0, stream>>>(resi, hs_t, NT);
    ln_kernel<<<gl(MT), 256, 0, stream>>>(resi, df[i].n1w, df[i].n1b, tka, MT);
    ln_kernel<<<gl(MT), 256, 0, stream>>>(tka, df[i].n1w, df[i].n1b, tkb, MT);  // double-LN per reference
    ln_kernel<<<gl(MT), 256, 0, stream>>>(pan_t, df[i].n2w, df[i].n2b, tkc, MT);
    gemm(tkc, df[i].m.Win2, nullptr, x2l, MT, 64, 32, 32, 64, 0, 0);
    dwconv1d_silu_kernel<<<gl(Bn*L*64), 256, 0, stream>>>(x2l, 64, df[i].m.conv2w, df[i].m.conv2b, x2c, Bn, L, 64);
    mamba_core(df[i].m, tkb, x2c, tka);
    tok2nchw_kernel<<<gl(2*32*HW), 256, 0, stream>>>(tka, t0, Bn, 32, HW);
    dw3x3_kernel<<<gl(2*32*HW), 256, 0, stream>>>(t0, df[i].dww, df[i].dwb, t1, Bn, 32, 64, 64);
    nchw2tok_kernel<<<gl(2*32*HW), 256, 0, stream>>>(t1, hs_t, Bn, 32, HW);
  }

  // refine + final residual
  tok2nchw_kernel<<<gl(2*32*HW), 256, 0, stream>>>(hs_t, t0, Bn, 32, HW);
  conv3(t0, ciw, cib, nullptr, hs_f, 32, 32, 64, 64, 0);       // out0
  conv3(hs_f, p1w, p1b, nullptr, pan_f, 32, 32, 64, 64, 4);    // relu
  conv3(pan_f, p2w, p2b, nullptr, t1, 32, 32, 64, 64, 0);      // res
  meanhw_kernel<<<dim3(32, Bn), 256, 0, stream>>>(t1, ym, 32, HW);
  fc_kernel<<<1, 64, 0, stream>>>(ym, d1w, d1b, z1b, Bn, 32, 8, 4);
  fc_kernel<<<1, 64, 0, stream>>>(z1b, d2w, d2b, zatt, Bn, 8, 32, 5);
  scale_add_kernel<<<gl(2*32*HW), 256, 0, stream>>>(hs_f, t1, zatt, Bn, 32, HW);
  conv3(hs_f, clw, clb, hs_bic, (float*)d_out, 32, 31, 64, 64, 0);
}
